// KnnInvariantPointAttention_2430951489768
// MI455X (gfx1250) — matrix operand order repaired and checked by reading, not compiled, not run on hardware
//
#include <hip/hip_runtime.h>

// ---------------- problem constants ----------------
#define BB 4
#define NN 1024
#define KK 32
#define CS 384
#define CZ 128
#define CH 16
#define HH 12
#define PQ 4
#define PV 8
#define BN   (BB*NN)      // 4096
#define BNK  (BN*KK)      // 131072
#define PROJ_N 1152       // 192 q + 192 k + 192 v + 144 q_pts + 432 kv_pts
#define ZOUT_N 48         // 12 b_bias + 32 z_down + 4 pad
#define FEAT_N 960        // 192 o + 288 o_pt + 96 dists + 384 o_pair
#define OUT_N  384
#define INF_   100000.0f
#define EPS_   1e-8f
#define LN_EPS_ 1e-5f

typedef __attribute__((ext_vector_type(16))) _Float16 v16h;
typedef __attribute__((ext_vector_type(8)))  float    v8f;
typedef _Float16 f16;

// ---------------- LayerNorm rows -> f16 ----------------
// one wave32 per row; C <= 384 so each lane caches <=12 values
__global__ void ln_rows(const float* __restrict__ src, f16* __restrict__ dst,
                        const float* __restrict__ scale, const float* __restrict__ bias,
                        int rows, int C) {
  int wave = (int)((blockIdx.x * blockDim.x + threadIdx.x) >> 5);
  int lane = threadIdx.x & 31;
  if (wave >= rows) return;
  const float* r = src + (size_t)wave * C;
  float vals[12];
  int nv = 0;
  float s = 0.f;
  for (int c = lane; c < C; c += 32) { float v = r[c]; vals[nv++] = v; s += v; }
  for (int off = 16; off > 0; off >>= 1) s += __shfl_xor(s, off, 32);
  float mean = s / (float)C;
  float vs = 0.f;
  for (int i = 0; i < nv; ++i) { float d = vals[i] - mean; vs += d * d; }
  for (int off = 16; off > 0; off >>= 1) vs += __shfl_xor(vs, off, 32);
  float rstd = rsqrtf(vs / (float)C + LN_EPS_);
  f16* o = dst + (size_t)wave * C;
  nv = 0;
  for (int c = lane; c < C; c += 32) {
    float v = (vals[nv++] - mean) * rstd * scale[c] + bias[c];
    o[c] = (f16)v;
  }
}

// ---------------- weight convert + transpose: dst[n*pitch+k] = (f16)src[k*Nc+n] ----------
__global__ void convT(const float* __restrict__ src, f16* __restrict__ dst,
                      int Kd, int Nc, int pitch) {
  int total = Kd * Nc;
  for (int i = blockIdx.x * blockDim.x + threadIdx.x; i < total; i += gridDim.x * blockDim.x) {
    int k = i / Nc, n = i % Nc;
    dst[(size_t)n * pitch + k] = (f16)src[i];
  }
}

__global__ void fill_f16(f16* dst, int n) {
  for (int i = blockIdx.x * blockDim.x + threadIdx.x; i < n; i += gridDim.x * blockDim.x)
    dst[i] = (f16)0.f;
}

// ---------------- WMMA GEMM: C(MxN,f32) = A(MxK,f16 row-major) * Bt(NxK,f16) ---------
// register-blocked: each wave32 owns a 16 x (16*NT) strip; one A fragment per K-step
// feeds NT v_wmma_f32_16x16x32_f16 instructions (A-fragment reuse, higher WMMA density).
template <int NT>
__global__ void __launch_bounds__(256) wmma_gemm_nt(const f16* __restrict__ A,
                                                    const f16* __restrict__ Bt,
                                                    float* __restrict__ C,
                                                    int M, int Kd, int Nout) {
  int wave = (int)((blockIdx.x * blockDim.x + threadIdx.x) >> 5);
  int lane = threadIdx.x & 31;
  int nstrips = Nout / (16 * NT);
  int tm = wave / nstrips, ts = wave % nstrips;
  if (tm >= (M >> 4)) return;
  int hsel = lane >> 4;                                  // half-wave select
  const f16* ap = A + (size_t)((tm << 4) + (lane & 15)) * Kd + 16 * hsel;
  const f16* bp[NT];
#pragma unroll
  for (int n = 0; n < NT; ++n)
    bp[n] = Bt + (size_t)(ts * NT * 16 + n * 16 + (lane & 15)) * Kd + 16 * hsel;

  v8f acc[NT];
#pragma unroll
  for (int n = 0; n < NT; ++n) acc[n] = (v8f){};

  for (int kk = 0; kk < Kd; kk += 32) {
    v16h a;
#pragma unroll
    for (int i = 0; i < 16; ++i) a[i] = ap[kk + i];          // K = kk..kk+15 (+16 for hi half)
#pragma unroll
    for (int n = 0; n < NT; ++n) {
      v16h b;
#pragma unroll
      for (int i = 0; i < 16; ++i) b[i] = bp[n][kk + i];
      acc[n] = __builtin_amdgcn_wmma_f32_16x16x32_f16(false, a, false, b, (short)0,
                                                      acc[n], false, false);
    }
  }
  int crow0 = (tm << 4) + 8 * hsel;
#pragma unroll
  for (int n = 0; n < NT; ++n) {
    int ccol = ts * NT * 16 + n * 16 + (lane & 15);
#pragma unroll
    for (int r = 0; r < 8; ++r)
      C[(size_t)(crow0 + r) * Nout + ccol] = acc[n][r];
  }
}

// ---------------- KNN invariant point attention (one block per (b,n)) ----------------
__global__ void __launch_bounds__(256) ipa_attn(
    const float* __restrict__ proj,   // (BN, 1152)
    const float* __restrict__ zout,   // (BNK, 48): [0,12) b_bias, [12,44) z_down
    const int*   __restrict__ edge,   // (BN, K)
    const float* __restrict__ rot,    // (BN, 9)
    const float* __restrict__ trans,  // (BN, 3)
    const float* __restrict__ smask,  // (BN)
    const float* __restrict__ headw,  // (12)
    f16* __restrict__ feat)           // (BN, 960) f16
{
  __shared__ float sh_q[192];
  __shared__ float sh_qpts[144];      // (h,p,3) rotated + trans
  __shared__ float sh_bb[384];        // (kk,h)
  __shared__ float sh_a[384];         // (kk,h) logits -> probs
  __shared__ float sh_hw[12];
  __shared__ float sh_R[9];
  __shared__ float sh_t[3];
  __shared__ float sh_mask[32];
  __shared__ int   sh_idx[32];
  __shared__ float sh_kbuf[16 * 192]; // staged k / v rows (half of neighbors)
  __shared__ float sh_pbuf[16 * 288]; // staged k_pts / v_pts (rotated); reused for o_pt
  __shared__ float sh_zd[16 * 32];    // staged z_down rows

  const int t = threadIdx.x;
  const int row = blockIdx.x;         // b*N + n
  const int b = row / NN;
  const float* prow = proj + (size_t)row * PROJ_N;

  if (t < 9)               sh_R[t]      = rot[(size_t)row * 9 + t];
  if (t >= 9 && t < 12)    sh_t[t - 9]  = trans[(size_t)row * 3 + (t - 9)];
  if (t >= 32 && t < 44) {
    int h = t - 32;
    float x = headw[h];
    float sp = (x > 20.f) ? x : log1pf(expf(x));
    sh_hw[h] = sp * 0.13608276348795434f;          // sqrt(1/(3*PQ*9/2)) = sqrt(1/54)
  }
  if (t >= 64 && t < 96) {
    int kk = t - 64;
    int j = edge[(size_t)row * KK + kk];
    sh_idx[kk] = j;
    sh_mask[kk] = smask[row] * smask[b * NN + j];
  }
  for (int c = t; c < 192; c += 256) sh_q[c] = prow[c];
  for (int e = t; e < 384; e += 256) {
    int kk = e / 12, h = e % 12;
    sh_bb[e] = zout[((size_t)row * KK + kk) * ZOUT_N + h];
  }
  __syncthreads();
  for (int e = t; e < 144; e += 256) {             // q_pts: R*p + t
    int pe = e / 3, i = e % 3;
    float x = prow[576 + pe * 3 + 0], y = prow[576 + pe * 3 + 1], z = prow[576 + pe * 3 + 2];
    sh_qpts[e] = sh_R[i * 3 + 0] * x + sh_R[i * 3 + 1] * y + sh_R[i * 3 + 2] * z + sh_t[i];
  }
  __syncthreads();

  const float qk_scale = 0.14433756729740643f;     // 1/sqrt(3*C_H)
  const float b_scale  = 0.57735026918962584f;     // sqrt(1/3)

  // ---- logits, two halves of 16 neighbors ----
  for (int ph = 0; ph < 2; ++ph) {
    for (int e = t; e < 16 * 192; e += 256) {      // k rows
      int l = e / 192, c = e % 192;
      int j = sh_idx[ph * 16 + l];
      sh_kbuf[e] = proj[((size_t)(b * NN + j)) * PROJ_N + 192 + c];
    }
    for (int e = t; e < 16 * 144; e += 256) {      // k_pts, rotated in center frame
      int l = e / 144, f = e % 144;
      int pe = f / 3, i = f % 3;                   // pe = h*4+p
      int h = pe >> 2, p = pe & 3;
      int j = sh_idx[ph * 16 + l];
      const float* src = proj + ((size_t)(b * NN + j)) * PROJ_N + 720 + (h * 12 + p) * 3;
      sh_pbuf[l * 144 + f] = sh_R[i * 3 + 0] * src[0] + sh_R[i * 3 + 1] * src[1] +
                             sh_R[i * 3 + 2] * src[2] + sh_t[i];
    }
    __syncthreads();
    for (int e = t; e < 192; e += 256) {           // 16 neighbors x 12 heads
      int l = e / 12, h = e % 12, kk = ph * 16 + l;
      float dot = 0.f;
#pragma unroll
      for (int c = 0; c < 16; ++c) dot += sh_q[h * 16 + c] * sh_kbuf[l * 192 + h * 16 + c];
      float av = dot * qk_scale + b_scale * sh_bb[kk * 12 + h];
      float sq = 0.f;
#pragma unroll
      for (int p = 0; p < 4; ++p)
#pragma unroll
        for (int i = 0; i < 3; ++i) {
          float d = sh_qpts[(h * 4 + p) * 3 + i] - sh_pbuf[l * 144 + (h * 4 + p) * 3 + i];
          sq += d * d;
        }
      av -= 0.5f * sh_hw[h] * sq;
      av += INF_ * (sh_mask[kk] - 1.0f);
      sh_a[kk * 12 + h] = av;
    }
    __syncthreads();
  }

  // ---- softmax over K per head ----
  if (t < 12) {
    float m = -3.4e38f;
    for (int kk = 0; kk < 32; ++kk) m = fmaxf(m, sh_a[kk * 12 + t]);
    float s = 0.f;
    for (int kk = 0; kk < 32; ++kk) s += expf(sh_a[kk * 12 + t] - m);
    float inv = 1.f / s;
    for (int kk = 0; kk < 32; ++kk) sh_a[kk * 12 + t] = expf(sh_a[kk * 12 + t] - m) * inv;
  }
  __syncthreads();

  // ---- output accumulation: 864 entries (192 o, 288 o_pt raw, 384 o_pair) ----
  float acc[4] = {0.f, 0.f, 0.f, 0.f};
  for (int ph = 0; ph < 2; ++ph) {
    for (int e = t; e < 16 * 192; e += 256) {      // v rows
      int l = e / 192, c = e % 192;
      int j = sh_idx[ph * 16 + l];
      sh_kbuf[e] = proj[((size_t)(b * NN + j)) * PROJ_N + 384 + c];
    }
    for (int e = t; e < 16 * 288; e += 256) {      // v_pts rotated
      int l = e / 288, f = e % 288;
      int pe = f / 3, i = f % 3;                   // pe = h*8+pv
      int h = pe >> 3, p = 4 + (pe & 7);
      int j = sh_idx[ph * 16 + l];
      const float* src = proj + ((size_t)(b * NN + j)) * PROJ_N + 720 + (h * 12 + p) * 3;
      sh_pbuf[l * 288 + f] = sh_R[i * 3 + 0] * src[0] + sh_R[i * 3 + 1] * src[1] +
                             sh_R[i * 3 + 2] * src[2] + sh_t[i];
    }
    for (int e = t; e < 16 * 32; e += 256) {       // z_down rows
      int l = e / 32, c = e % 32;
      sh_zd[e] = zout[((size_t)row * KK + ph * 16 + l) * ZOUT_N + 12 + c];
    }
    __syncthreads();
#pragma unroll
    for (int r = 0; r < 4; ++r) {
      int id = t + 256 * r;
      if (id >= 864) break;
      float s = 0.f;
      if (id < 192) {                              // o[h][c]
        int h = id >> 4, c = id & 15;
        for (int l = 0; l < 16; ++l)
          s += sh_a[(ph * 16 + l) * 12 + h] * sh_kbuf[l * 192 + h * 16 + c];
      } else if (id < 480) {                       // o_pt raw
        int e = id - 192;
        int h = (e / 3) >> 3;
        for (int l = 0; l < 16; ++l)
          s += sh_a[(ph * 16 + l) * 12 + h] * sh_pbuf[l * 288 + e];
      } else {                                     // o_pair[h][c]
        int e = id - 480;
        int h = e >> 5, c = e & 31;
        for (int l = 0; l < 16; ++l)
          s += sh_a[(ph * 16 + l) * 12 + h] * sh_zd[l * 32 + c];
      }
      acc[r] += s;
    }
    __syncthreads();
  }

  // ---- write features: [o | o_pt | dists | o_pair] ----
  f16* frow = feat + (size_t)row * FEAT_N;
#pragma unroll
  for (int r = 0; r < 4; ++r) {
    int id = t + 256 * r;
    if (id >= 864) break;
    float v = acc[r];
    if (id < 192)        frow[id] = (f16)v;
    else if (id < 480)   sh_pbuf[id - 192] = v;            // stash raw o_pt
    else                 frow[576 + (id - 480)] = (f16)v;  // o_pair
  }
  __syncthreads();
  for (int e = t; e < 288; e += 256) {             // inverse frame: R^T (o_pt - t)
    int pe = e / 3, i = e % 3;
    float x = sh_pbuf[pe * 3 + 0] - sh_t[0];
    float y = sh_pbuf[pe * 3 + 1] - sh_t[1];
    float z = sh_pbuf[pe * 3 + 2] - sh_t[2];
    float v = sh_R[0 * 3 + i] * x + sh_R[1 * 3 + i] * y + sh_R[2 * 3 + i] * z;
    sh_pbuf[512 + e] = v;
    frow[192 + e] = (f16)v;
  }
  __syncthreads();
  for (int d = t; d < 96; d += 256) {
    float x = sh_pbuf[512 + d * 3 + 0];
    float y = sh_pbuf[512 + d * 3 + 1];
    float z = sh_pbuf[512 + d * 3 + 2];
    frow[480 + d] = (f16)sqrtf(x * x + y * y + z * z + EPS_);
  }
}

// ---------------- launch ----------------
extern "C" void kernel_launch(void* const* d_in, const int* in_sizes, int n_in,
                              void* d_out, int out_size, void* d_ws, size_t ws_size,
                              hipStream_t stream) {
  const float* s      = (const float*)d_in[0];
  const float* z      = (const float*)d_in[1];
  const int*   edge   = (const int*)  d_in[2];
  const float* rot    = (const float*)d_in[3];
  const float* trans  = (const float*)d_in[4];
  const float* smask  = (const float*)d_in[5];
  const float* ln_s_s = (const float*)d_in[6];
  const float* ln_s_b = (const float*)d_in[7];
  const float* ln_z_s = (const float*)d_in[8];
  const float* ln_z_b = (const float*)d_in[9];
  const float* w_q    = (const float*)d_in[10];
  const float* w_k    = (const float*)d_in[11];
  const float* w_v    = (const float*)d_in[12];
  const float* w_qp   = (const float*)d_in[13];
  const float* w_kvp  = (const float*)d_in[14];
  const float* w_b    = (const float*)d_in[15];
  const float* w_dz   = (const float*)d_in[16];
  const float* headw  = (const float*)d_in[17];
  const float* w_out  = (const float*)d_in[18];
  float* out = (float*)d_out;

  // workspace layout (256B aligned)
  char* ws = (char*)d_ws;
  size_t off = 0;
  auto alloc = [&](size_t bytes) -> void* {
    void* p = ws + off;
    off = (off + bytes + 255) & ~(size_t)255;
    return p;
  };
  f16*   s_ln  = (f16*)  alloc((size_t)BN  * CS * sizeof(f16));
  f16*   z_ln  = (f16*)  alloc((size_t)BNK * CZ * sizeof(f16));
  f16*   Wt    = (f16*)  alloc((size_t)PROJ_N * CS * sizeof(f16));
  f16*   Wz    = (f16*)  alloc((size_t)ZOUT_N * CZ * sizeof(f16));
  f16*   Wo    = (f16*)  alloc((size_t)OUT_N * FEAT_N * sizeof(f16));
  float* proj  = (float*)alloc((size_t)BN  * PROJ_N * sizeof(float));
  float* zoutb = (float*)alloc((size_t)BNK * ZOUT_N * sizeof(float));
  f16*   feats = (f16*)  alloc((size_t)BN  * FEAT_N * sizeof(f16));

  // 1) LayerNorms -> f16
  ln_rows<<<BN / 8, 256, 0, stream>>>(s, s_ln, ln_s_s, ln_s_b, BN, CS);
  ln_rows<<<BNK / 8, 256, 0, stream>>>(z, z_ln, ln_z_s, ln_z_b, BNK, CZ);

  // 2) weights -> f16, transposed (column-major for WMMA B fragments)
  fill_f16<<<8, 256, 0, stream>>>(Wz, ZOUT_N * CZ);            // zero pad cols 44..47
  convT<<<128, 256, 0, stream>>>(w_q,   Wt + (size_t)0   * CS, CS, 192, CS);
  convT<<<128, 256, 0, stream>>>(w_k,   Wt + (size_t)192 * CS, CS, 192, CS);
  convT<<<128, 256, 0, stream>>>(w_v,   Wt + (size_t)384 * CS, CS, 192, CS);
  convT<<<128, 256, 0, stream>>>(w_qp,  Wt + (size_t)576 * CS, CS, 144, CS);
  convT<<<128, 256, 0, stream>>>(w_kvp, Wt + (size_t)720 * CS, CS, 432, CS);
  convT<<<8,   256, 0, stream>>>(w_b,   Wz + (size_t)0  * CZ,  CZ, HH, CZ);
  convT<<<8,   256, 0, stream>>>(w_dz,  Wz + (size_t)12 * CZ,  CZ, 32, CZ);
  convT<<<256, 256, 0, stream>>>(w_out, Wo, FEAT_N, OUT_N, FEAT_N);

  // 3) WMMA GEMMs
  // proj: 4096x384 @ 384x1152 ; strips of 64 cols -> (4096/16)*(1152/64) waves
  wmma_gemm_nt<4><<<(BN / 16) * (PROJ_N / 64) / 8, 256, 0, stream>>>(
      s_ln, Wt, proj, BN, CS, PROJ_N);
  // z-path: 131072x128 @ 128x48 ; strips of 48 cols (NT=3)
  wmma_gemm_nt<3><<<(BNK / 16) * (ZOUT_N / 48) / 8, 256, 0, stream>>>(
      z_ln, Wz, zoutb, BNK, CZ, ZOUT_N);

  // 4) attention
  ipa_attn<<<BN, 256, 0, stream>>>(proj, zoutb, edge, rot, trans, smask, headw, feats);

  // 5) final WMMA GEMM -> d_out (fp32): 4096x960 @ 960x384 ; strips of 64 cols
  wmma_gemm_nt<4><<<(BN / 16) * (OUT_N / 64) / 8, 256, 0, stream>>>(
      feats, Wo, out, BN, FEAT_N, OUT_N);
}
